// DTDRMiniTransformer_88441966559316
// MI455X (gfx1250) — matrix operand order repaired and checked by reading, not compiled, not run on hardware
//
#include <hip/hip_runtime.h>
#include <hip/hip_bf16.h>
#include <math.h>

// ---------------------------------------------------------------------------
// DTDR mini-transformer block for MI455X (gfx1250, wave32, WMMA).
// GEMMs: V_WMMA_F32_16X16X32_F16 with double-buffered LDS tiles staged via
// GLOBAL_LOAD_ASYNC_TO_LDS_B128 (ASYNCcnt). Weights are pre-dequantized
// int8(int32)->f16 once (exact), so the GEMM hot loop is copy + WMMA only.
// ---------------------------------------------------------------------------

typedef __attribute__((ext_vector_type(16))) _Float16 v16h;
typedef __attribute__((ext_vector_type(4)))  _Float16 v4h;
typedef __attribute__((ext_vector_type(8)))  float    v8f;

#define D_MODEL 2048
#define N_HEADS 16
#define D_HEAD  128
#define D_FF    8192
#define BATCH   4
#define SEQ     1024
#define MTOK    (BATCH * SEQ)   // 4096 token rows

static __device__ inline v8f wmma_f16(v16h a, v16h b, v8f c) {
    return __builtin_amdgcn_wmma_f32_16x16x32_f16(
        /*neg_a=*/false, a, /*neg_b=*/false, b,
        /*c_mod=*/(short)0, c, /*reuse_a=*/false, /*reuse_b=*/false);
}

// A-matrix 16x32 f16 fragment (ISA 7.12.2): lane holds row (lane&15); halfs
// j=0..7 come from K = 8*(lane>>4)+j, j=8..15 from K = 16+8*(lane>>4)+(j-8).
static __device__ inline v16h load_afrag(const _Float16* p, int stride) {
    int lane = threadIdx.x & 31;
    const _Float16* q = p + (size_t)(lane & 15) * stride + ((lane >> 4) << 3);
    union { v16h v; uint4 u[2]; } r;
    r.u[0] = *(const uint4*)q;
    r.u[1] = *(const uint4*)(q + 16);
    return r.v;
}

// B-matrix 32x16 f16 fragment: lane holds column N=(lane&15); halfs j=0..7 from
// K = 8*(lane>>4)+j, j=8..15 from K = 16+8*(lane>>4)+(j-8). Source is row-major W[N][K].
static __device__ inline v16h load_bfrag(const _Float16* p, int stride) {
    int lane = threadIdx.x & 31;
    const _Float16* q = p + (size_t)(lane & 15) * stride + ((lane >> 4) << 3);
    union { v16h v; uint4 u[2]; } r;
    r.u[0] = *(const uint4*)q;
    r.u[1] = *(const uint4*)(q + 16);
    return r.v;
}

// Async global -> LDS 16-byte copy (CDNA5, tracked by ASYNCcnt). The LDS
// destination is the raw LDS byte offset (low 32 bits of the generic pointer).
static __device__ inline void async_copy_b128(void* lds_ptr, const void* gptr) {
    unsigned int loff = (unsigned int)(uintptr_t)lds_ptr;
    unsigned long long ga = (unsigned long long)(uintptr_t)gptr;
    asm volatile("global_load_async_to_lds_b128 %0, %1, off"
                 :: "v"(loff), "v"(ga) : "memory");
}
static __device__ inline void wait_asynccnt0() {
    asm volatile("s_wait_asynccnt 0" ::: "memory");
}

// ---------------------------------------------------------------------------
// int8 (delivered as int32) -> f16 weight dequant, exact for |q| <= 127.
// Each thread converts one int4 (4 values).
// ---------------------------------------------------------------------------
__global__ __launch_bounds__(256)
void dequant_w_kernel(const int* __restrict__ in, _Float16* __restrict__ out) {
    size_t idx = (size_t)blockIdx.x * 256 + threadIdx.x;
    int4 v = ((const int4*)in)[idx];
    v4h o;
    o.x = (_Float16)(float)v.x;
    o.y = (_Float16)(float)v.y;
    o.z = (_Float16)(float)v.z;
    o.w = (_Float16)(float)v.w;
    *(v4h*)(out + idx * 4) = o;
}

// ---------------------------------------------------------------------------
// LayerNorm + orthonormal Hadamard (n = 2048) fused, f32 in -> f16 out.
// ---------------------------------------------------------------------------
__global__ __launch_bounds__(256)
void ln_hadamard_kernel(const float* __restrict__ x, const float* __restrict__ g,
                        const float* __restrict__ bvec, _Float16* __restrict__ out) {
    __shared__ float buf[D_MODEL];
    __shared__ float red[16];
    const int tid = threadIdx.x;
    const int lane = tid & 31, wv = tid >> 5;
    const size_t row = blockIdx.x;
    const float* xr = x + row * D_MODEL;

    float s0 = 0.f, s1 = 0.f;
    for (int i = tid; i < D_MODEL; i += 256) {
        float v = xr[i]; buf[i] = v; s0 += v; s1 += v * v;
    }
    for (int off = 16; off; off >>= 1) {
        s0 += __shfl_xor(s0, off, 32);
        s1 += __shfl_xor(s1, off, 32);
    }
    if (lane == 0) { red[wv] = s0; red[8 + wv] = s1; }
    __syncthreads();
    if (tid == 0) {
        float a = 0.f, b2 = 0.f;
        for (int i = 0; i < 8; ++i) { a += red[i]; b2 += red[8 + i]; }
        float mu = a / (float)D_MODEL;
        float var = b2 / (float)D_MODEL - mu * mu;
        red[0] = mu;
        red[1] = rsqrtf(var + 1e-5f);
    }
    __syncthreads();
    float mu = red[0], rs = red[1];
    for (int i = tid; i < D_MODEL; i += 256)
        buf[i] = (buf[i] - mu) * rs * g[i] + bvec[i];
    __syncthreads();
    for (int hh = 1; hh < D_MODEL; hh <<= 1) {
        for (int p = tid; p < D_MODEL / 2; p += 256) {
            int i = ((p & ~(hh - 1)) << 1) | (p & (hh - 1));
            float a = buf[i], c = buf[i + hh];
            buf[i] = a + c; buf[i + hh] = a - c;
        }
        __syncthreads();
    }
    const float rn = 0.02209708691207961f; // 1/sqrt(2048)
    for (int i = tid; i < D_MODEL; i += 256)
        out[row * D_MODEL + i] = (_Float16)(buf[i] * rn);
}

// ---------------------------------------------------------------------------
// In-place orthonormal Hadamard on f16 rows (n = 2048 or 8192).
// ---------------------------------------------------------------------------
__global__ __launch_bounds__(256)
void hadamard_f16_kernel(_Float16* __restrict__ data, int n, float rn) {
    __shared__ float buf[D_FF];
    const int tid = threadIdx.x;
    _Float16* r = data + (size_t)blockIdx.x * n;
    for (int i = tid; i < n; i += 256) buf[i] = (float)r[i];
    __syncthreads();
    for (int hh = 1; hh < n; hh <<= 1) {
        for (int p = tid; p < n / 2; p += 256) {
            int i = ((p & ~(hh - 1)) << 1) | (p & (hh - 1));
            float a = buf[i], c = buf[i + hh];
            buf[i] = a + c; buf[i + hh] = a - c;
        }
        __syncthreads();
    }
    for (int i = tid; i < n; i += 256) r[i] = (_Float16)(buf[i] * rn);
}

// ---------------------------------------------------------------------------
// WMMA GEMM: Y[M,N] = A[M,K](f16) x Wf[N,K](f16) * s[N]; 128x128 tile per
// 256-thread workgroup; 8 waves in 4x2, each 32x64; BK=32. Tiles are staged
// with async global->LDS copies, double-buffered to overlap with WMMA.
// MODE 0: y*s -> f16 [B,H,T,D_HEAD]         (QKV, head remap)
// MODE 1: res + y*s -> f32 [M,N]            (attn out projection + residual)
// MODE 2: gelu(y*s + bias) -> f16 [M,N]     (FFN1)
// MODE 3: res + y*s + bias -> f32 [M,N]     (FFN2 + residual = final output)
// ---------------------------------------------------------------------------
#define TSTR 40   // LDS tile row stride in halfs (80 B, 16B-aligned rows)

static __device__ inline void stage_tiles_async(
    const _Float16* __restrict__ A, const _Float16* __restrict__ Wf,
    int K, int tileM, int tileN, int kk,
    _Float16* asbuf, _Float16* bsbuf, int tid) {
    // 512 16-byte chunks per matrix (128 rows x 4 chunks), 2 per thread each.
#pragma unroll
    for (int r = 0; r < 2; ++r) {
        int c = tid + r * 256;
        int row = c >> 2, ch = c & 3;
        async_copy_b128(asbuf + row * TSTR + ch * 8,
                        A + (size_t)(tileM + row) * K + kk + ch * 8);
        async_copy_b128(bsbuf + row * TSTR + ch * 8,
                        Wf + (size_t)(tileN + row) * K + kk + ch * 8);
    }
}

template <int MODE>
__global__ __launch_bounds__(256)
void gemm_f16_wmma_kernel(const _Float16* __restrict__ A, const _Float16* __restrict__ Wf,
                          const float* __restrict__ s, const float* __restrict__ bias,
                          const float* __restrict__ res, void* __restrict__ dst,
                          int N, int K) {
    __shared__ _Float16 As[2][128 * TSTR];
    __shared__ _Float16 Bs[2][128 * TSTR];

    const int tid = threadIdx.x;
    const int lane = tid & 31;
    const int wv = tid >> 5;
    const int wy = wv >> 1;  // 0..3 -> M sub-tile
    const int wx = wv & 1;   // 0..1 -> N sub-tile
    const int tileM = blockIdx.y * 128;
    const int tileN = blockIdx.x * 128;

    v8f acc[2][4];
#pragma unroll
    for (int a = 0; a < 2; ++a)
#pragma unroll
        for (int b = 0; b < 4; ++b) acc[a][b] = (v8f)0.f;

    // Prologue: stage tile 0.
    stage_tiles_async(A, Wf, K, tileM, tileN, 0, As[0], Bs[0], tid);
    wait_asynccnt0();
    __syncthreads();

    int sel = 0;
    for (int kk = 0; kk < K; kk += 32) {
        // Overlap: issue async copies for the next tile while computing.
        if (kk + 32 < K)
            stage_tiles_async(A, Wf, K, tileM, tileN, kk + 32,
                              As[sel ^ 1], Bs[sel ^ 1], tid);

        v16h a0 = load_afrag(As[sel] + (wy * 32 + 0) * TSTR, TSTR);
        v16h a1 = load_afrag(As[sel] + (wy * 32 + 16) * TSTR, TSTR);
#pragma unroll
        for (int nt = 0; nt < 4; ++nt) {
            v16h bf = load_bfrag(Bs[sel] + (wx * 64 + nt * 16) * TSTR, TSTR);
            acc[0][nt] = wmma_f16(a0, bf, acc[0][nt]);
            acc[1][nt] = wmma_f16(a1, bf, acc[1][nt]);
        }

        // Own async copies done, then block-wide visibility/ordering.
        wait_asynccnt0();
        __syncthreads();
        sel ^= 1;
    }

    // Epilogue. C/D layout: lane holds col (lane&15), rows (lane>>4)*8 + i.
#pragma unroll
    for (int mt = 0; mt < 2; ++mt)
#pragma unroll
        for (int nt = 0; nt < 4; ++nt)
#pragma unroll
            for (int i = 0; i < 8; ++i) {
                int gm = tileM + wy * 32 + mt * 16 + ((lane >> 4) << 3) + i;
                int gn = tileN + wx * 64 + nt * 16 + (lane & 15);
                float v = acc[mt][nt][i] * s[gn];
                if (MODE == 0) {
                    int b_ = gm >> 10, t_ = gm & 1023;
                    int hh = gn >> 7, dd = gn & 127;
                    ((_Float16*)dst)[(((size_t)(b_ * N_HEADS + hh) * SEQ) + t_) * D_HEAD + dd] =
                        (_Float16)v;
                } else if (MODE == 1) {
                    size_t o = (size_t)gm * N + gn;
                    ((float*)dst)[o] = res[o] + v;
                } else if (MODE == 2) {
                    v += bias[gn];
                    v = 0.5f * v * (1.0f + erff(v * 0.70710678118654752f));
                    ((_Float16*)dst)[(size_t)gm * N + gn] = (_Float16)v;
                } else {
                    size_t o = (size_t)gm * N + gn;
                    ((float*)dst)[o] = res[o] + v + bias[gn];
                }
            }
}

// ---------------------------------------------------------------------------
// Flash-style attention (unchanged from round 1): one workgroup = one (b,h)
// x 256 query rows; Q fragments in registers; score B-frags straight from
// global k; V staged transposed in LDS; P through per-wave LDS scratch with
// an explicit s_wait_dscnt for the wave-local RAW; online softmax.
// ---------------------------------------------------------------------------
__global__ __launch_bounds__(256)
void attention_kernel(const _Float16* __restrict__ q, const _Float16* __restrict__ k,
                      const _Float16* __restrict__ v, _Float16* __restrict__ ao) {
    __shared__ _Float16 Vt[128 * 72];      // V^T: [d][key], stride 72 halfs
    __shared__ _Float16 Ps[8][32 * 64];    // per-wave P tile, stride 64 halfs

    const int tid = threadIdx.x;
    const int lane = tid & 31;
    const int wave = tid >> 5;
    const int bh = blockIdx.x;             // 0..63
    const int b_ = bh >> 4, h_ = bh & 15;
    const int qrow0 = blockIdx.y * 256 + wave * 32;
    const size_t bhbase = (size_t)bh * SEQ * D_HEAD;
    const float SCALE = 0.08838834764831845f; // 1/sqrt(128)

    v16h qa[2][4];
#pragma unroll
    for (int mt = 0; mt < 2; ++mt)
#pragma unroll
        for (int ks = 0; ks < 4; ++ks)
            qa[mt][ks] = load_afrag(q + bhbase + (size_t)(qrow0 + mt * 16) * D_HEAD + ks * 32,
                                    D_HEAD);

    v8f o[2][8];
#pragma unroll
    for (int mt = 0; mt < 2; ++mt)
#pragma unroll
        for (int nt = 0; nt < 8; ++nt) o[mt][nt] = (v8f)0.f;
    float mrow[2][8], lrow[2][8];
#pragma unroll
    for (int mt = 0; mt < 2; ++mt)
#pragma unroll
        for (int i = 0; i < 8; ++i) { mrow[mt][i] = -INFINITY; lrow[mt][i] = 0.f; }

    for (int kb = 0; kb < SEQ / 64; ++kb) {
        __syncthreads();
        for (int idx = tid; idx < 64 * 128; idx += 256) {
            int key = idx >> 7, d = idx & 127;
            Vt[d * 72 + key] = v[bhbase + (size_t)(kb * 64 + key) * D_HEAD + d];
        }
        __syncthreads();

        v8f sacc[2][4];
#pragma unroll
        for (int mt = 0; mt < 2; ++mt)
#pragma unroll
            for (int n4 = 0; n4 < 4; ++n4) sacc[mt][n4] = (v8f)0.f;
#pragma unroll
        for (int n4 = 0; n4 < 4; ++n4)
#pragma unroll
            for (int ks = 0; ks < 4; ++ks) {
                v16h bf = load_bfrag(
                    k + bhbase + (size_t)(kb * 64 + n4 * 16) * D_HEAD + ks * 32, D_HEAD);
                sacc[0][n4] = wmma_f16(qa[0][ks], bf, sacc[0][n4]);
                sacc[1][n4] = wmma_f16(qa[1][ks], bf, sacc[1][n4]);
            }

#pragma unroll
        for (int mt = 0; mt < 2; ++mt)
#pragma unroll
            for (int i = 0; i < 8; ++i) {
                float mx = -INFINITY;
#pragma unroll
                for (int n4 = 0; n4 < 4; ++n4)
                    mx = fmaxf(mx, sacc[mt][n4][i] * SCALE);
                for (int off = 1; off < 16; off <<= 1)
                    mx = fmaxf(mx, __shfl_xor(mx, off, 32));
                float newm = fmaxf(mrow[mt][i], mx);
                float fac = __expf(mrow[mt][i] - newm);
                float ls = 0.f;
                int prow = mt * 16 + ((lane >> 4) << 3) + i;
#pragma unroll
                for (int n4 = 0; n4 < 4; ++n4) {
                    float p = __expf(sacc[mt][n4][i] * SCALE - newm);
                    ls += p;
                    Ps[wave][prow * 64 + n4 * 16 + (lane & 15)] = (_Float16)p;
                }
                for (int off = 1; off < 16; off <<= 1)
                    ls += __shfl_xor(ls, off, 32);
                lrow[mt][i] = lrow[mt][i] * fac + ls;
                mrow[mt][i] = newm;
#pragma unroll
                for (int nt = 0; nt < 8; ++nt) o[mt][nt][i] *= fac;
            }

        asm volatile("s_wait_dscnt 0" ::: "memory");

        v16h pa[2][2];
#pragma unroll
        for (int mt = 0; mt < 2; ++mt)
#pragma unroll
            for (int k2 = 0; k2 < 2; ++k2)
                pa[mt][k2] = load_afrag(&Ps[wave][(mt * 16) * 64 + k2 * 32], 64);
#pragma unroll
        for (int nt = 0; nt < 8; ++nt)
#pragma unroll
            for (int k2 = 0; k2 < 2; ++k2) {
                v16h bf = load_bfrag(&Vt[(nt * 16) * 72 + k2 * 32], 72);
                o[0][nt] = wmma_f16(pa[0][k2], bf, o[0][nt]);
                o[1][nt] = wmma_f16(pa[1][k2], bf, o[1][nt]);
            }
    }

#pragma unroll
    for (int mt = 0; mt < 2; ++mt)
#pragma unroll
        for (int i = 0; i < 8; ++i) {
            float linv = 1.0f / lrow[mt][i];
            int t_ = qrow0 + mt * 16 + ((lane >> 4) << 3) + i;
#pragma unroll
            for (int nt = 0; nt < 8; ++nt) {
                int dd = nt * 16 + (lane & 15);
                ao[((size_t)b_ * SEQ + t_) * D_MODEL + h_ * D_HEAD + dd] =
                    (_Float16)(o[mt][nt][i] * linv);
            }
        }
}

// ---------------------------------------------------------------------------
// Host-side orchestration
// ---------------------------------------------------------------------------
extern "C" void kernel_launch(void* const* d_in, const int* in_sizes, int n_in,
                              void* d_out, int out_size, void* d_ws, size_t ws_size,
                              hipStream_t stream) {
    (void)in_sizes; (void)n_in; (void)out_size; (void)ws_size;
    const float* x    = (const float*)d_in[0];
    const float* g1   = (const float*)d_in[1];
    const float* b1   = (const float*)d_in[2];
    const float* g2   = (const float*)d_in[3];
    const float* b2   = (const float*)d_in[4];
    const int*   Qq   = (const int*)d_in[5];   const float* sq  = (const float*)d_in[6];
    const int*   Qk   = (const int*)d_in[7];   const float* sk  = (const float*)d_in[8];
    const int*   Qv   = (const int*)d_in[9];   const float* sv  = (const float*)d_in[10];
    const int*   Qo   = (const int*)d_in[11];  const float* so  = (const float*)d_in[12];
    const int*   Qf1  = (const int*)d_in[13];  const float* sf1 = (const float*)d_in[14];
    const float* bf1  = (const float*)d_in[15];
    const int*   Qf2  = (const int*)d_in[16];  const float* sf2 = (const float*)d_in[17];
    const float* bf2  = (const float*)d_in[18];

    char* w = (char*)d_ws;
    const size_t MB = (size_t)1 << 20;
    _Float16* h1  = (_Float16*)(w + 0 * MB);     // 16 MB  f16 [4096,2048]
    _Float16* qb  = (_Float16*)(w + 16 * MB);    // 16 MB  f16 [B,H,T,128]
    _Float16* kb  = (_Float16*)(w + 32 * MB);    // 16 MB
    _Float16* vb  = (_Float16*)(w + 48 * MB);    // 16 MB
    _Float16* ao  = (_Float16*)(w + 64 * MB);    // 16 MB  f16 [4096,2048]
    float*    x1  = (float*)   (w + 80 * MB);    // 32 MB  f32 [4096,2048]
    _Float16* h2  = (_Float16*)(w + 112 * MB);   // 16 MB  f16 [4096,2048]
    _Float16* ff  = (_Float16*)(w + 128 * MB);   // 64 MB  f16 [4096,8192]
    _Float16* wq  = (_Float16*)(w + 192 * MB);   //  8 MB  f16 [2048,2048]
    _Float16* wk  = (_Float16*)(w + 200 * MB);   //  8 MB
    _Float16* wvv = (_Float16*)(w + 208 * MB);   //  8 MB
    _Float16* wo  = (_Float16*)(w + 216 * MB);   //  8 MB
    _Float16* wf1 = (_Float16*)(w + 224 * MB);   // 32 MB  f16 [8192,2048]
    _Float16* wf2 = (_Float16*)(w + 256 * MB);   // 32 MB  f16 [2048,8192]

    const dim3 blk(256);
    const dim3 gN2048(D_MODEL / 128, MTOK / 128);   // (16,32)
    const dim3 gN8192(D_FF / 128,  MTOK / 128);     // (64,32)
    const float rn2048 = 0.02209708691207961f;      // 1/sqrt(2048)
    const float rn8192 = 0.011048543456039806f;     // 1/sqrt(8192)
    const int NDM2 = D_MODEL * D_MODEL;             // 4 Mi elements
    const int NFF  = D_MODEL * D_FF;                // 16 Mi elements

    // One-time (per launch) exact weight dequant int8->f16.
    dequant_w_kernel<<<NDM2 / 1024, blk, 0, stream>>>(Qq,  wq);
    dequant_w_kernel<<<NDM2 / 1024, blk, 0, stream>>>(Qk,  wk);
    dequant_w_kernel<<<NDM2 / 1024, blk, 0, stream>>>(Qv,  wvv);
    dequant_w_kernel<<<NDM2 / 1024, blk, 0, stream>>>(Qo,  wo);
    dequant_w_kernel<<<NFF  / 1024, blk, 0, stream>>>(Qf1, wf1);
    dequant_w_kernel<<<NFF  / 1024, blk, 0, stream>>>(Qf2, wf2);

    // h1 = hadamard(LN1(x))
    ln_hadamard_kernel<<<MTOK, blk, 0, stream>>>(x, g1, b1, h1);
    // q,k,v projections
    gemm_f16_wmma_kernel<0><<<gN2048, blk, 0, stream>>>(h1, wq,  sq, nullptr, nullptr, qb, D_MODEL, D_MODEL);
    gemm_f16_wmma_kernel<0><<<gN2048, blk, 0, stream>>>(h1, wk,  sk, nullptr, nullptr, kb, D_MODEL, D_MODEL);
    gemm_f16_wmma_kernel<0><<<gN2048, blk, 0, stream>>>(h1, wvv, sv, nullptr, nullptr, vb, D_MODEL, D_MODEL);
    // attention -> ao [B,T,2048]
    attention_kernel<<<dim3(BATCH * N_HEADS, SEQ / 256), blk, 0, stream>>>(qb, kb, vb, ao);
    // hadamard(ao), then x1 = x + ao_h @ Wo^T * so
    hadamard_f16_kernel<<<MTOK, blk, 0, stream>>>(ao, D_MODEL, rn2048);
    gemm_f16_wmma_kernel<1><<<gN2048, blk, 0, stream>>>(ao, wo, so, nullptr, x, x1, D_MODEL, D_MODEL);
    // h2 = hadamard(LN2(x1))
    ln_hadamard_kernel<<<MTOK, blk, 0, stream>>>(x1, g2, b2, h2);
    // ff = gelu(h2 @ Wf1^T * sf1 + bf1)
    gemm_f16_wmma_kernel<2><<<gN8192, blk, 0, stream>>>(h2, wf1, sf1, bf1, nullptr, ff, D_FF, D_MODEL);
    // hadamard(ff) over 8192, then out = x1 + ff_h @ Wf2^T * sf2 + bf2
    hadamard_f16_kernel<<<MTOK, blk, 0, stream>>>(ff, D_FF, rn8192);
    gemm_f16_wmma_kernel<3><<<gN2048, blk, 0, stream>>>(ff, wf2, sf2, bf2, x1, d_out, D_MODEL, D_FF);
}
